// DynamicFineMatching_44581760532730
// MI455X (gfx1250) — matrix operand order repaired and checked by reading, not compiled, not run on hardware
//
#include <hip/hip_runtime.h>

typedef __attribute__((ext_vector_type(16))) _Float16     v16h;
typedef __attribute__((ext_vector_type(2)))  _Float16     h2;
typedef __attribute__((ext_vector_type(8)))  float        v8f;
typedef __attribute__((ext_vector_type(4)))  float        f32x4;
typedef __attribute__((ext_vector_type(8)))  unsigned int v8u;

#define WW   25
#define WDIM 5
#define CCH  128

__device__ __forceinline__ float wave_sum(float v){
#pragma unroll
  for (int o = 16; o > 0; o >>= 1) v += __shfl_xor(v, o, 32);
  return v;
}
__device__ __forceinline__ float wave_max(float v){
#pragma unroll
  for (int o = 16; o > 0; o >>= 1) v = fmaxf(v, __shfl_xor(v, o, 32));
  return v;
}
// pack two f32 into one dword of two f16 (v_cvt_pk_f16_f32)
__device__ __forceinline__ unsigned pack2(float a, float b){
  h2 h; h[0] = (_Float16)a; h[1] = (_Float16)b;
  return __builtin_bit_cast(unsigned, h);
}
// Load 16 consecutive f32 channels of one F1 row, convert to the 16-half
// B-register block (VGPR v = K=2v,2v+1 for this lane's column).
__device__ __forceinline__ v16h load_row_f16(const float* __restrict__ rowp){
  const f32x4* q = (const f32x4*)rowp;
  f32x4 x0 = __builtin_nontemporal_load(&q[0]);
  f32x4 x1 = __builtin_nontemporal_load(&q[1]);
  f32x4 x2 = __builtin_nontemporal_load(&q[2]);
  f32x4 x3 = __builtin_nontemporal_load(&q[3]);
  v8u r;
  r[0] = pack2(x0[0], x0[1]);  r[1] = pack2(x0[2], x0[3]);
  r[2] = pack2(x1[0], x1[1]);  r[3] = pack2(x1[2], x1[3]);
  r[4] = pack2(x2[0], x2[1]);  r[5] = pack2(x2[2], x2[3]);
  r[6] = pack2(x3[0], x3[1]);  r[7] = pack2(x3[2], x3[3]);
  return __builtin_bit_cast(v16h, r);
}

__global__ __launch_bounds__(256) void fine_matching_kernel(
    const float* __restrict__ feat_f0,
    const float* __restrict__ feat_f1,
    const float* __restrict__ score0,
    const float* __restrict__ mconf_in,
    const float* __restrict__ mk0c,
    const float* __restrict__ mk1c,
    const unsigned char* __restrict__ gt_mask,
    float* __restrict__ out,
    int M)
{
  const int lane = (int)(threadIdx.x & 31u);
  const int m    = (int)((blockIdx.x * blockDim.x + threadIdx.x) >> 5);
  if (m >= M) return;   // uniform per wave: EXEC stays all-ones for WMMA

  const float* s0p = score0  + (size_t)m * WW;
  const float* f0p = feat_f0 + (size_t)m * (WW * CCH);
  const float* f1p = feat_f1 + (size_t)m * (WW * CCH);

  // Hint target-window features toward the caches (global_prefetch_b8).
  __builtin_prefetch(f1p + lane * 100, 0, 0);

  // ---- softmax(score/TEMP) over the 25-lane reference window ----
  float t0    = (lane < WW) ? s0p[lane] * 10.0f : -__builtin_inff();
  float t0max = wave_max(t0);
  float e0    = (lane < WW) ? __expf(t0 - t0max) : 0.0f;
  float e0sum = wave_sum(e0);
  float h0    = e0 / e0sum;              // lane r holds heatmap0[r] (0 for r>=25)

  // soft-argmax grid: linspace(-1,1,5); row-major r = h*5 + w
  float gx  = -1.0f + 0.5f * (float)(lane % WDIM);
  float gy  = -1.0f + 0.5f * (float)(lane / WDIM);
  float c0x = wave_sum(h0 * gx);
  float c0y = wave_sum(h0 * gy);

  // ---- weighted pool p[c] = sum_r h0[r]*F0[r,c]; 4 channels per lane ----
  const f32x4* F0 = (const f32x4*)f0p;
  f32x4 p = {0.f, 0.f, 0.f, 0.f};
#pragma unroll 5
  for (int r = 0; r < WW; ++r){
    float w  = __shfl(h0, r, 32);
    f32x4 f  = __builtin_nontemporal_load(&F0[r * 32 + lane]);  // stream-once
    p += w * f;
  }

  // ---- sim[r] = F1[r,:]·p via v_wmma_f32_16x16x32_f16 ----
  // A = p broadcast into all 16 rows; B columns = F1 rows (two tiles cover
  // r=0..24).  D rows are then identical, and lane l holds column l&15, so
  // sim falls out with no post-shuffles at all.
  const int hi  = lane >> 4;
  const int col = lane & 15;          // B column owned by this lane
  const int hi8 = hi * 8;
  const unsigned pk01 = pack2(p[0], p[1]);   // p channels (4L, 4L+1)
  const unsigned pk23 = pack2(p[2], p[3]);   // p channels (4L+2, 4L+3)

  v8f acc0 = {};
  v8f acc1 = {};
#pragma unroll
  for (int k = 0; k < 4; ++k){
    const int c0 = 32 * k;

    // A (16x32 f16, rows identical): lo lanes need K{0..15}, hi lanes
    // K{16..31}; each packed pair lives whole in one owner lane.
    v8u araw;
#pragma unroll
    for (int v = 0; v < 8; ++v){
      const int ch = c0 + hi * 16 + 2 * v;
      const unsigned src = (v & 1) ? pk23 : pk01;
      araw[v] = (unsigned)__shfl((int)src, ch >> 2, 32);   // ds_bpermute
    }
    const v16h amat = __builtin_bit_cast(v16h, araw);

    // B tile 0: column `col` = F1 row `col`, 64B contiguous per lane.
    const v16h b0 = load_row_f16(f1p + col * CCH + c0 + hi * 16);

    // B tile 1: rows 16..24 (columns 9..15 zero-padded).
    v16h b1 = {};
    const int row1 = 16 + col;
    if (row1 < WW)
      b1 = load_row_f16(f1p + row1 * CCH + c0 + hi * 16);

    acc0 = __builtin_amdgcn_wmma_f32_16x16x32_f16(false, amat, false, b0,
                                                  (short)0, acc0, false, false);
    acc1 = __builtin_amdgcn_wmma_f32_16x16x32_f16(false, amat, false, b1,
                                                  (short)0, acc1, false, false);
  }

  // lane r<16: tile0 column r; lane r>=16: tile1 column r-16 => window row r.
  float sim = ((lane < 16) ? acc0[0] : acc1[0]) * 0.08838834764831845f; // 1/sqrt(128)

  // ---- second softmax, conf, coords ----
  float t1    = (lane < WW) ? sim : -__builtin_inff();
  float t1max = wave_max(t1);
  float e1    = (lane < WW) ? __expf(t1 - t1max) : 0.0f;
  float e1sum = wave_sum(e1);
  float h1    = e1 / e1sum;
  float c1x   = wave_sum(h1 * gx);
  float c1y   = wave_sum(h1 * gy);
  float mconf = (1.0f / e1sum) * mconf_in[m];  // max(softmax) = 1/sum(shifted exp)

  float keep = gt_mask[m] ? 0.0f : 1.0f;
  if (lane < 5){
    float res;                                 // half*SCALE = 2*8 = 16
    if      (lane == 0) res = mk0c[2 * m + 0] + c0x * 16.0f;
    else if (lane == 1) res = mk0c[2 * m + 1] + c0y * 16.0f;
    else if (lane == 2) res = mk1c[2 * m + 0] + c1x * 16.0f;
    else if (lane == 3) res = mk1c[2 * m + 1] + c1y * 16.0f;
    else                res = mconf;
    out[(size_t)m * 5 + lane] = res * keep;
  }
}

extern "C" void kernel_launch(void* const* d_in, const int* in_sizes, int n_in,
                              void* d_out, int out_size, void* d_ws, size_t ws_size,
                              hipStream_t stream) {
  const float*         feat_f0 = (const float*)d_in[0];
  const float*         feat_f1 = (const float*)d_in[1];
  const float*         score0  = (const float*)d_in[2];
  const float*         mconf   = (const float*)d_in[3];
  const float*         mk0     = (const float*)d_in[4];
  const float*         mk1     = (const float*)d_in[5];
  const unsigned char* gt      = (const unsigned char*)d_in[6];
  float*               out     = (float*)d_out;

  const int M = in_sizes[3];                 // mconf_in is [M]
  const int waves_per_block = 256 / 32;
  const int blocks = (M + waves_per_block - 1) / waves_per_block;
  fine_matching_kernel<<<dim3(blocks), dim3(256), 0, stream>>>(
      feat_f0, feat_f1, score0, mconf, mk0, mk1, gt, out, M);
}
